// KANSNN_15831249453625
// MI455X (gfx1250) — compile-verified, matrix operand order checked by reading, not run on hardware
//
#include <hip/hip_runtime.h>
#include <stdint.h>

// ---------------------------------------------------------------------------
// KAN-SNN fused forward for gfx1250 (MI455X).
//
//  B=256, T=250, N_IN=700, N_HID=512, N_OUT=20
//  Combined per-step GEMM: i1 = [feat(16x2816), s1(16x512)] x [Wcat ; w_rec]
//  -> 104 k-tiles of 32 (bf16 WMMA, f32 accum).
//
//  Grid: 16 blocks (one per 16 batch rows), 512 threads (16 waves),
//  each wave owns 2 N-tiles (16x16) of the hidden dim for all 250 steps.
//  x rows for step t are staged to LDS via async global->LDS copies
//  (GLOBAL_LOAD_ASYNC_TO_LDS_B128, tracked with ASYNCcnt).
// ---------------------------------------------------------------------------

#define B_TOT   256
#define T_STEPS 250
#define N_IN    700
#define N_HID   512
#define N_OUT   20

#define KFEAT     2800
#define FKT       88            // feature k-tiles (2816 / 32)
#define RKT       16            // recurrent k-tiles (512 / 32)
#define TKT       (FKT + RKT)   // 104
#define NTILES    (N_HID / 16)  // 32

#define XROW_STRIDE 704                   // padded row stride (16B aligned)
#define AFRAG_DW  (TKT * 32 * 8)          // 26624 dwords = 106496 B
#define S1_FLOATS (16 * N_HID)            // 8192 floats  =  32768 B
#define XSTAGE_FLOATS (16 * XROW_STRIDE)  // 11264 floats =  45056 B
#define SMEM_BYTES ((AFRAG_DW + S1_FLOATS + XSTAGE_FLOATS) * 4)  // 184320 B

#define BPACK_DW  (TKT * NTILES * 32 * 8) // 851968 dwords (~3.25 MB in d_ws)

typedef __bf16 bf16_t;
typedef bf16_t v16bf __attribute__((ext_vector_type(16)));
typedef float  v8f   __attribute__((ext_vector_type(8)));
typedef int    v4i_raw __attribute__((vector_size(16)));   // builtin's pointee type

union Frag {
    uint4 q[2];
    v16bf v;
};

#if defined(__gfx1250__) && \
    __has_builtin(__builtin_amdgcn_global_load_async_to_lds_b128) && \
    __has_builtin(__builtin_amdgcn_s_wait_asynccnt)
#define USE_ASYNC_LDS 1
#endif

__device__ __forceinline__ uint16_t f32_to_bf16(float f) {
    uint32_t u = __float_as_uint(f);
    uint32_t r = u + 0x7FFFu + ((u >> 16) & 1u);   // round-to-nearest-even
    return (uint16_t)(r >> 16);
}

// ISA 16-bit A/B fragment layout (16x32, wave32):
//  lane<16  holds K in {0..7} U {16..23}; lane>=16 holds K in {8..15} U {24..31}
//  dword j<4: K = 2j+half (+laneHi*8); j>=4: K = 16 + 2(j-4)+half (+laneHi*8)
__device__ __forceinline__ int kin_of(int lane, int j, int half) {
    int base = (lane >= 16) ? 8 : 0;
    int k = (j < 4) ? (2 * j + half) : (16 + 2 * (j - 4) + half);
    return k + base;
}

// ---------------------------------------------------------------------------
// Pack kernel: build bf16 B fragments for the combined [Wcat ; w_rec] matrix.
// Layout: Bpack[kt][nt][lane][8 dwords]
// ---------------------------------------------------------------------------
__global__ __launch_bounds__(256) void kansnn_pack_b(
    const float* __restrict__ w_kan, const float* __restrict__ d1,
    const float* __restrict__ d2,    const float* __restrict__ d3,
    const float* __restrict__ w_rec, uint32_t* __restrict__ Bpack)
{
    int idx = blockIdx.x * blockDim.x + threadIdx.x;
    if (idx >= BPACK_DW) return;
    int j    = idx & 7;
    int lane = (idx >> 3) & 31;
    int nt   = (idx >> 8) & 31;
    int kt   = idx >> 13;
    int n    = nt * 16 + (lane & 15);   // hidden unit (B column)

    uint32_t pack = 0;
#pragma unroll
    for (int h = 0; h < 2; ++h) {
        int k_in = kin_of(lane, j, h);
        float v = 0.0f;
        if (kt < FKT) {
            int kg = kt * 32 + k_in;
            if (kg < KFEAT) {
                int seg, kf;
                if      (kg <  700) { seg = 0; kf = kg;        }
                else if (kg < 1400) { seg = 1; kf = kg -  700; }
                else if (kg < 2100) { seg = 2; kf = kg - 1400; }
                else                { seg = 3; kf = kg - 2100; }
                const float* w = (seg == 0) ? w_kan : (seg == 1) ? d1
                               : (seg == 2) ? d2    : d3;
                v = w[n * N_IN + kf];
            }
        } else {
            int k = (kt - FKT) * 32 + k_in;          // hidden index
            v = w_rec[n * N_HID + k];                // (s1 @ w_rec^T)[.,n]
        }
        pack |= ((uint32_t)f32_to_bf16(v)) << (16 * h);
    }
    Bpack[idx] = pack;
}

// ---------------------------------------------------------------------------
// Fused scan kernel: one block = 16 batch rows for all 250 timesteps.
// ---------------------------------------------------------------------------
__global__ __launch_bounds__(512) void kansnn_scan(
    const float* __restrict__ x,          // [B, T, N_IN]
    const uint32_t* __restrict__ Bpack,   // packed bf16 fragments
    const float* __restrict__ w_out,      // [N_OUT, N_HID] f32
    float* __restrict__ out)              // [B, N_OUT]
{
    extern __shared__ uint32_t smem[];
    uint32_t* afrag    = smem;                                   // [TKT][32][8]
    float*    s1_plain = (float*)(smem + AFRAG_DW);              // [16][N_HID]
    float*    xstage   = (float*)(smem + AFRAG_DW + S1_FLOATS);  // [16][704]

    const int tid  = threadIdx.x;
    const int lane = tid & 31;
    const int wv   = tid >> 5;              // 0..15
    const int b0   = blockIdx.x * 16;

    const int nt0   = wv * 2;
    const int nt1   = wv * 2 + 1;
    const int ncol  = lane & 15;            // N within tile
    const int mbase = (lane >> 4) * 8;      // D layout: lanes>=16 hold M=8..15

    // Persistent LIF state, register-resident (fixed (m,n) ownership per lane).
    float v1s[2][8], a1s[2][8], s1s[2][8];
#pragma unroll
    for (int i = 0; i < 2; ++i)
#pragma unroll
        for (int r = 0; r < 8; ++r) { v1s[i][r] = 0.f; a1s[i][r] = 0.f; s1s[i][r] = 0.f; }

    // Readout state (threads 0..319 own one (m, o) pair).
    const int ro_m = tid / N_OUT;
    const int ro_o = tid - ro_m * N_OUT;
    float vout = 0.f, accout = 0.f;

    // Zero the recurrent s1 fragment region and the spike plane.
    for (int i = tid; i < RKT * 32 * 8; i += 512) afrag[FKT * 32 * 8 + i] = 0u;
    for (int i = tid; i < S1_FLOATS;    i += 512) s1_plain[i] = 0.f;
    __syncthreads();

    const uint4* bp4 = (const uint4*)Bpack;

    for (int t = 0; t < T_STEPS; ++t) {
        // ---- 0. Stage x rows for this step into LDS (async -> LDS) --------
        // 16 rows x 700 f32 = 2800 float4 transfers.
        for (int idx = tid; idx < 16 * (N_IN / 4); idx += 512) {
            int row = idx / (N_IN / 4);
            int col = idx - row * (N_IN / 4);
            const float* gsrc = x + ((size_t)(b0 + row) * T_STEPS + t) * N_IN + col * 4;
            float* ldst = xstage + row * XROW_STRIDE + col * 4;
#if defined(USE_ASYNC_LDS)
            __builtin_amdgcn_global_load_async_to_lds_b128(
                (__attribute__((address_space(1))) v4i_raw*)gsrc,
                (__attribute__((address_space(3))) v4i_raw*)ldst, 0, 0);
#else
            *(float4*)ldst = *(const float4*)gsrc;
#endif
        }
#if defined(USE_ASYNC_LDS)
        __builtin_amdgcn_s_wait_asynccnt(0);
#endif
        __syncthreads();   // xstage ready

        // ---- 1. KAN feature fragments -> LDS (bf16 A layout) --------------
        for (int idx = tid; idx < FKT * 32 * 8; idx += 512) {
            int j  = idx & 7;
            int ln = (idx >> 3) & 31;
            int kt = idx >> 8;
            int m  = ln & 15;
            const float* xrow = xstage + m * XROW_STRIDE;
            uint32_t pack = 0;
#pragma unroll
            for (int h = 0; h < 2; ++h) {
                int kg = kt * 32 + kin_of(ln, j, h);
                float v = 0.0f;
                if (kg < KFEAT) {
                    int seg, kf;
                    if      (kg <  700) { seg = 0; kf = kg;        }
                    else if (kg < 1400) { seg = 1; kf = kg -  700; }
                    else if (kg < 2100) { seg = 2; kf = kg - 1400; }
                    else                { seg = 3; kf = kg - 2100; }
                    float xv = xrow[kf];
                    float t1 = fminf(fabsf(xv), 1.0f);
                    v = (seg == 0) ? xv : (seg == 1) ? t1
                      : (seg == 2) ? t1 * t1 : t1 * t1 * t1;
                }
                pack |= ((uint32_t)f32_to_bf16(v)) << (16 * h);
            }
            afrag[idx] = pack;
        }
        __syncthreads();   // fragments ready

        // ---- 2. Combined GEMM: i1 = [feat, s1] x [Wcat ; w_rec] -----------
        v8f c0 = {};
        v8f c1 = {};
        for (int kt = 0; kt < TKT; ++kt) {
            if (kt + 1 < TKT)
                __builtin_prefetch(bp4 + (size_t)(((kt + 1) * NTILES + nt0) * 32 + lane) * 2, 0, 3);

            Frag a;
            const uint4* ap = (const uint4*)(afrag + ((kt * 32) + lane) * 8);
            a.q[0] = ap[0];
            a.q[1] = ap[1];

            Frag b0f;
            const uint4* bq0 = bp4 + (size_t)((kt * NTILES + nt0) * 32 + lane) * 2;
            b0f.q[0] = bq0[0];
            b0f.q[1] = bq0[1];
            c0 = __builtin_amdgcn_wmma_f32_16x16x32_bf16(
                    false, a.v, false, b0f.v, (short)0, c0, false, false);

            Frag b1f;
            const uint4* bq1 = bp4 + (size_t)((kt * NTILES + nt1) * 32 + lane) * 2;
            b1f.q[0] = bq1[0];
            b1f.q[1] = bq1[1];
            c1 = __builtin_amdgcn_wmma_f32_16x16x32_bf16(
                    false, a.v, false, b1f.v, (short)0, c1, false, false);
        }
        __syncthreads();   // all afrag reads done before s1 fragment rewrite

        // ---- 3. Adaptive-LIF update + scatter new spikes ------------------
        auto do_update = [&](const v8f& c, int i, int nt) {
#pragma unroll
            for (int r = 0; r < 8; ++r) {
                float i1 = c[r];
                float sp = s1s[i][r];
                float v  = 0.95f * v1s[i][r] + 0.05f * i1 - 1.0f * sp;
                float a  = 0.85f * a1s[i][r] + 0.15f * sp;
                float sn = ((v - (1.0f + 0.05f * a)) > 0.0f) ? 1.0f : 0.0f;
                v1s[i][r] = v; a1s[i][r] = a; s1s[i][r] = sn;

                int m = mbase + r;
                int n = nt * 16 + ncol;
                s1_plain[m * N_HID + n] = sn;

                // place s1(m, k=n) into bf16 A-fragment layout for next step
                int kt  = FKT + (n >> 5);
                int kin = n & 31;
                int lt  = m + ((kin & 8) ? 16 : 0);
                int jj  = (kin < 16) ? ((kin & 7) >> 1) : (4 + ((kin & 7) >> 1));
                int hf  = kin & 1;
                ((uint16_t*)afrag)[(((kt * 32 + lt) * 8) + jj) * 2 + hf] =
                    (sn != 0.0f) ? (uint16_t)0x3F80 : (uint16_t)0;
            }
        };
        do_update(c0, 0, nt0);
        do_update(c1, 1, nt1);
        __syncthreads();   // spike plane + fragments ready

        // ---- 4. Readout LIF (20 outputs per row) --------------------------
        if (tid < 16 * N_OUT) {
            const float* wo = w_out + ro_o * N_HID;
            const float* sp = s1_plain + ro_m * N_HID;
            float io = 0.f;
#pragma unroll 4
            for (int k = 0; k < N_HID; k += 4) {
                io += sp[k]     * wo[k]
                    + sp[k + 1] * wo[k + 1]
                    + sp[k + 2] * wo[k + 2]
                    + sp[k + 3] * wo[k + 3];
            }
            vout = 0.9f * vout + io;
            float so = (vout - 1.0f > 0.0f) ? 1.0f : 0.0f;
            vout -= so;              // THRESH = 1
            accout += vout;
        }
        __syncthreads();   // readout done before s1_plain / xstage rewrite
    }

    if (tid < 16 * N_OUT)
        out[(b0 + ro_m) * N_OUT + ro_o] = accout * (1.0f / (float)T_STEPS);
}

// ---------------------------------------------------------------------------
extern "C" void kernel_launch(void* const* d_in, const int* in_sizes, int n_in,
                              void* d_out, int out_size, void* d_ws, size_t ws_size,
                              hipStream_t stream)
{
    const float* x     = (const float*)d_in[0];
    const float* w_kan = (const float*)d_in[1];
    const float* d1    = (const float*)d_in[2];
    const float* d2    = (const float*)d_in[3];
    const float* d3    = (const float*)d_in[4];
    const float* w_rec = (const float*)d_in[5];
    const float* w_out = (const float*)d_in[6];
    float* out = (float*)d_out;

    uint32_t* Bpack = (uint32_t*)d_ws;   // BPACK_DW dwords (~3.25 MB)

    kansnn_pack_b<<<BPACK_DW / 256, 256, 0, stream>>>(w_kan, d1, d2, d3, w_rec, Bpack);

    kansnn_scan<<<B_TOT / 16, 512, SMEM_BYTES, stream>>>(x, Bpack, w_out, out);
}
